// BaseLDPU_17162689315027
// MI455X (gfx1250) — matrix operand order repaired and checked by reading, not compiled, not run on hardware
//
#include <hip/hip_runtime.h>
#include <math.h>

// ---------------- types ----------------
typedef __attribute__((ext_vector_type(16))) __bf16 bf16x16;
typedef __attribute__((ext_vector_type(8)))  __bf16 bf16x8;
typedef __attribute__((ext_vector_type(8)))  float  f32x8;

static constexpr int kF    = 100;
static constexpr int kO    = 20;
static constexpr int kPair = 2000;   // kF*kO
static constexpr int kD    = 1936;
static constexpr int kH    = 8;
static constexpr int kDH   = 242;
static constexpr int kDFF  = 2048;
static constexpr int kFeat = 2376;
static constexpr int LDSK  = 40;     // padded bf16 K stride (80B rows, 16B aligned)

struct ConvGeom { int icn, ihn, iwn, ohn, own, khn, kwn, stride, pad; };

__device__ __forceinline__ unsigned pack_bf16(float a, float b)
{
    unsigned short lo = __builtin_bit_cast(unsigned short, (__bf16)a);
    unsigned short hi = __builtin_bit_cast(unsigned short, (__bf16)b);
    return ((unsigned)hi << 16) | (unsigned)lo;
}

// ---------------------------------------------------------------------------
// Generic GEMM:  C[m,n] = act(sum_k A(m,k) * W[n,k] + bias[n]) * bnsc[n] + bnsh[n] + res[m,n]
// AMODE (compile-time):
//   0: A[m*lda + k*akstride]
//   1: A[aidx[2m+aidxsel]*lda + k]                (row gather, pair_idx; akstride==1)
//   2: A[(m/49)*lda + (m%49) + k*akstride]        (union NCHW 1x1-conv)
//   3: im2col, input NCHW  (conv1 over spatial_masks)
//   4: im2col, input NHWC  (conv2 over pooled tmp)
// BF16 WMMA 16x16x32, f32 accumulation. 128x64 block tile, 8 wave32 waves,
// double-buffered LDS staging, per-block hoisted row descriptors.
// ---------------------------------------------------------------------------
template<int ACT, int AMODE>
__global__ __launch_bounds__(256)
void gemm_bf16_wmma(const float* __restrict__ A, const float* __restrict__ W,
                    const float* __restrict__ bias,
                    const float* __restrict__ bnsc, const float* __restrict__ bnsh,
                    const float* __restrict__ res, float* __restrict__ C,
                    int M, int N, int Kdim,
                    const int* __restrict__ aidx, int aidxsel,
                    long long lda, long long akstride, ConvGeom cg,
                    long long ldc, long long ldres)
{
    constexpr bool CONVM = (AMODE >= 3);

    __shared__ __align__(16) __bf16 As[2][128 * LDSK];
    __shared__ __align__(16) __bf16 Bs[2][64 * LDSK];
    __shared__ long long RB[128];     // per-row A base offset (-1 = row OOB)
    __shared__ int RH[CONVM ? 128 : 1], RW[CONVM ? 128 : 1];

    const int tid  = threadIdx.x;
    const int lane = tid & 31;
    const int wave = tid >> 5;
    const int wm   = wave & 3;              // 4 row-blocks of 32
    const int wn   = wave >> 2;             // 2 col-blocks of 32
    const long long bm = (long long)blockIdx.x * 128;
    const long long bn = (long long)blockIdx.y * 64;
    const int OHW = cg.ohn * cg.own;
    const int KHW = cg.khn * cg.kwn;

    // ---- per-row descriptors: all row-decode division leaves the K-loop ----
    for (int r = tid; r < 128; r += 256) {
        long long m = bm + r;
        long long base = -1;
        if (m < M) {
            if constexpr (AMODE == 0) {
                base = m * lda;
            } else if constexpr (AMODE == 1) {
                base = (long long)aidx[2 * (int)m + aidxsel] * lda;
            } else if constexpr (AMODE == 2) {
                base = (m / 49) * lda + (m % 49);
            } else {
                int mi  = (int)m;
                int img = mi / OHW, ohw = mi % OHW;
                int oh  = ohw / cg.own, ow = ohw % cg.own;
                base = (long long)img * cg.icn * cg.ihn * cg.iwn;
                RH[r] = oh * cg.stride - cg.pad;
                RW[r] = ow * cg.stride - cg.pad;
            }
        }
        RB[r] = base;
    }
    __syncthreads();

    f32x8 acc[2][2];
#pragma unroll
    for (int i = 0; i < 2; ++i)
#pragma unroll
        for (int j = 0; j < 2; ++j)
#pragma unroll
            for (int t = 0; t < 8; ++t) acc[i][j][t] = 0.f;

    const int kp = tid & 15;          // fixed k-pair column per thread

    // ---- tile staging (pair-packed bf16 stores, b64 global loads) ----
    auto stageA = [&](int buf, int k0) {
        const int k0e = k0 + 2 * kp;
        const int k1e = k0e + 1;
        if constexpr (!CONVM) {
            const long long koff0 = (long long)k0e * akstride;
            const long long koff1 = (long long)k1e * akstride;
            for (int r = tid >> 4; r < 128; r += 16) {
                float v0 = 0.f, v1 = 0.f;
                long long base = RB[r];
                if (base >= 0) {
                    if (akstride == 1 && k1e < Kdim) {
                        float2 t = *(const float2*)(A + base + k0e);
                        v0 = t.x; v1 = t.y;
                    } else {
                        if (k0e < Kdim) v0 = A[base + koff0];
                        if (k1e < Kdim) v1 = A[base + koff1];
                    }
                }
                *(unsigned*)&As[buf][r * LDSK + 2 * kp] = pack_bf16(v0, v1);
            }
        } else {
            int ic0 = k0e / KHW, kr0 = k0e % KHW;
            int ic1 = k1e / KHW, kr1 = k1e % KHW;
            int kh0 = kr0 / cg.kwn, kw0 = kr0 % cg.kwn;
            int kh1 = kr1 / cg.kwn, kw1 = kr1 % cg.kwn;
            long long co0 = (AMODE == 3) ? (long long)ic0 * cg.ihn * cg.iwn : (long long)ic0;
            long long co1 = (AMODE == 3) ? (long long)ic1 * cg.ihn * cg.iwn : (long long)ic1;
            for (int r = tid >> 4; r < 128; r += 16) {
                float v0 = 0.f, v1 = 0.f;
                long long base = RB[r];
                if (base >= 0) {
                    int ih0 = RH[r] + kh0, iw0 = RW[r] + kw0;
                    int ih1 = RH[r] + kh1, iw1 = RW[r] + kw1;
                    if (k0e < Kdim && ih0 >= 0 && ih0 < cg.ihn && iw0 >= 0 && iw0 < cg.iwn) {
                        long long o = (AMODE == 3)
                            ? base + co0 + (long long)ih0 * cg.iwn + iw0
                            : base + ((long long)ih0 * cg.iwn + iw0) * cg.icn + co0;
                        v0 = A[o];
                    }
                    if (k1e < Kdim && ih1 >= 0 && ih1 < cg.ihn && iw1 >= 0 && iw1 < cg.iwn) {
                        long long o = (AMODE == 3)
                            ? base + co1 + (long long)ih1 * cg.iwn + iw1
                            : base + ((long long)ih1 * cg.iwn + iw1) * cg.icn + co1;
                        v1 = A[o];
                    }
                }
                *(unsigned*)&As[buf][r * LDSK + 2 * kp] = pack_bf16(v0, v1);
            }
        }
    };
    auto stageB = [&](int buf, int k0) {
        const int k0e = k0 + 2 * kp;
        const int k1e = k0e + 1;
        for (int r = tid >> 4; r < 64; r += 16) {
            float v0 = 0.f, v1 = 0.f;
            long long n = bn + r;
            if (n < N) {
                const float* p = W + n * (long long)Kdim;
                if (k1e < Kdim) { float2 t = *(const float2*)(p + k0e); v0 = t.x; v1 = t.y; }
                else if (k0e < Kdim) v0 = p[k0e];
            }
            *(unsigned*)&Bs[buf][r * LDSK + 2 * kp] = pack_bf16(v0, v1);
        }
    };

    const int nsteps = (Kdim + 31) / 32;
    stageA(0, 0);
    stageB(0, 0);
    __syncthreads();

    const int r    = lane & 15;
    const int half = lane >> 4;

    for (int it = 0; it < nsteps; ++it) {
        const int cur = it & 1;
        if (it + 1 < nsteps) {                 // stage next tile into other buffer
            stageA(cur ^ 1, (it + 1) * 32);
            stageB(cur ^ 1, (it + 1) * 32);
            if (it + 2 < nsteps)               // hint step+2 weights -> global_prefetch_b8
                __builtin_prefetch(W + (bn + (tid & 63)) * (long long)Kdim + (it + 2) * 32, 0, 1);
        }
        // ---- WMMA on current buffer (fragments per ISA 16-bit layouts) ----
#pragma unroll
        for (int i = 0; i < 2; ++i) {
            const __bf16* ar = &As[cur][(wm * 32 + i * 16 + r) * LDSK];
            bf16x8 a0 = *(const bf16x8*)(ar + half * 8);        // K = half*8 .. +7
            bf16x8 a1 = *(const bf16x8*)(ar + 16 + half * 8);   // K = 16+half*8 .. +7
            bf16x16 afrag;
#pragma unroll
            for (int t = 0; t < 8; ++t) { afrag[t] = a0[t]; afrag[8 + t] = a1[t]; }
#pragma unroll
            for (int j = 0; j < 2; ++j) {
                const __bf16* br = &Bs[cur][(wn * 32 + j * 16 + r) * LDSK];
                bf16x8 b0 = *(const bf16x8*)(br + half * 8);        // K = half*8 .. +7
                bf16x8 b1 = *(const bf16x8*)(br + 16 + half * 8);   // K = 16+half*8 .. +7
                bf16x16 bfrag;
#pragma unroll
                for (int t = 0; t < 8; ++t) { bfrag[t] = b0[t]; bfrag[8 + t] = b1[t]; }
                acc[i][j] = __builtin_amdgcn_wmma_f32_16x16x32_bf16(
                    false, afrag, false, bfrag, (short)0, acc[i][j], false, false);
            }
        }
        __syncthreads();
    }

    // ---- epilogue (C layout: VGPR v, lanes0-15 M=v, lanes16-31 M=v+8) ----
    const int hi = lane >> 4;
#pragma unroll
    for (int i = 0; i < 2; ++i)
#pragma unroll
        for (int j = 0; j < 2; ++j)
#pragma unroll
            for (int v = 0; v < 8; ++v) {
                long long row = bm + wm * 32 + i * 16 + v + hi * 8;
                long long col = bn + wn * 32 + j * 16 + r;
                if (row < M && col < N) {
                    float x = acc[i][j][v];
                    if (bias) x += bias[col];
                    if (ACT == 1) x = fmaxf(x, 0.f);
                    if (ACT == 2) x = 1.f / (1.f + __expf(-x));
                    if (bnsc) x = x * bnsc[col] + bnsh[col];
                    if (res)  x += res[row * ldres + col];
                    C[row * ldc + col] = x;
                }
            }
}

// ---------------- attention (per b,h,q block) ----------------
__global__ __launch_bounds__(128)
void attn_kernel(const float* __restrict__ Q, const float* __restrict__ Kx,
                 const float* __restrict__ V, float* __restrict__ O,
                 int B, int S, int causal)
{
    const int bid = blockIdx.x;
    const int q = bid % S, h = (bid / S) % kH, b = bid / (S * kH);
    const int tid = threadIdx.x;
    __shared__ float sc[128];
    __shared__ float red[128];
    const long long tq = (long long)b * S + q;
    const float* qrow = Q + tq * kD + h * kDH;
    const int kmax = causal ? (q + 1) : S;
    const float scale = 0.0642824346533225f;  // 1/sqrt(242)

    for (int k = tid; k < S; k += 128) {
        float s = -__builtin_inff();
        if (k < kmax) {
            const float* krow = Kx + ((long long)b * S + k) * kD + h * kDH;
            float a = 0.f;
            for (int d = 0; d < kDH; ++d) a += qrow[d] * krow[d];
            s = a * scale;
        }
        sc[k] = s;
    }
    __syncthreads();
    float mx = -__builtin_inff();
    for (int k = tid; k < S; k += 128) mx = fmaxf(mx, sc[k]);
    red[tid] = mx; __syncthreads();
    for (int off = 64; off > 0; off >>= 1) {
        if (tid < off) red[tid] = fmaxf(red[tid], red[tid + off]);
        __syncthreads();
    }
    mx = red[0]; __syncthreads();
    float sm = 0.f;
    for (int k = tid; k < S; k += 128) { float e = __expf(sc[k] - mx); sc[k] = e; sm += e; }
    red[tid] = sm; __syncthreads();
    for (int off = 64; off > 0; off >>= 1) {
        if (tid < off) red[tid] += red[tid + off];
        __syncthreads();
    }
    const float inv = 1.f / red[0];
    __syncthreads();
    float* orow = O + tq * kD + h * kDH;
    for (int d = tid; d < kDH; d += 128) {
        float a = 0.f;
        for (int k = 0; k < kmax; ++k)
            a += sc[k] * V[((long long)b * S + k) * kD + h * kDH + d];
        orow[d] = a * inv;
    }
}

// ---------------- layernorm (block per row) ----------------
__global__ __launch_bounds__(256)
void ln_kernel(const float* __restrict__ X, const float* __restrict__ g,
               const float* __restrict__ b, float* __restrict__ Y, int Dn)
{
    const long long row = blockIdx.x;
    const int tid = threadIdx.x;
    const float* x = X + row * Dn;
    __shared__ float rs[256], rq[256];
    float s = 0.f, q = 0.f;
    for (int i = tid; i < Dn; i += 256) { float v = x[i]; s += v; q += v * v; }
    rs[tid] = s; rq[tid] = q; __syncthreads();
    for (int off = 128; off > 0; off >>= 1) {
        if (tid < off) { rs[tid] += rs[tid + off]; rq[tid] += rq[tid + off]; }
        __syncthreads();
    }
    const float mean = rs[0] / Dn;
    const float var  = rq[0] / Dn - mean * mean;
    const float inv  = rsqrtf(var + 1e-5f);
    for (int i = tid; i < Dn; i += 256)
        Y[row * Dn + i] = (x[i] - mean) * inv * g[i] + b[i];
}

// ---------------- misc elementwise kernels ----------------
__global__ void bn_prep_kernel(const float* g, const float* b, const float* m,
                               const float* v, float* sc, float* sh, int n)
{
    int i = blockIdx.x * 256 + threadIdx.x;
    if (i >= n) return;
    float inv = g[i] * rsqrtf(v[i] + 1e-5f);
    sc[i] = inv; sh[i] = b[i] - m[i] * inv;
}

__global__ void perm_vrw_kernel(const float* __restrict__ W, float* __restrict__ WP)
{
    long long idx = (long long)blockIdx.x * 256 + threadIdx.x;   // 512*12544
    if (idx >= 512LL * 12544) return;
    int kp = (int)(idx % 12544);      // k' = hw*256 + oc
    int n  = (int)(idx / 12544);
    int oc = kp % 256, hw = kp / 256;
    WP[idx] = W[(long long)n * 12544 + oc * 49 + hw];
}

__global__ void maxpool_kernel(const float* __restrict__ t1, float* __restrict__ t2)
{
    long long idx = (long long)blockIdx.x * 256 + threadIdx.x;   // 2000*49*128
    if (idx >= 2000LL * 49 * 128) return;
    int ic = (int)(idx % 128);
    long long r = idx / 128;
    int pw = (int)(r % 7), ph = (int)((r / 7) % 7);
    long long p = r / 49;
    float mx = -__builtin_inff();
    for (int dh = 0; dh < 3; ++dh) {
        int ih = ph * 2 - 1 + dh;
        if (ih < 0 || ih >= 14) continue;
        for (int dw = 0; dw < 3; ++dw) {
            int iw = pw * 2 - 1 + dw;
            if (iw < 0 || iw >= 14) continue;
            mx = fmaxf(mx, t1[((p * 196) + ih * 14 + iw) * 128 + ic]);
        }
    }
    t2[idx] = mx;
}

__global__ void xsem_kernel(const int* __restrict__ pair_idx, const int* __restrict__ lab,
                            const float* __restrict__ e1, const float* __restrict__ e2,
                            float* __restrict__ REL)
{
    long long idx = (long long)blockIdx.x * 256 + threadIdx.x;   // 2000*400
    if (idx >= 2000LL * 400) return;
    int j = (int)(idx % 400);
    int p = (int)(idx / 400);
    float v;
    if (j < 200) { int c = lab[pair_idx[2 * p]];     v = e1[c * 200 + j]; }
    else         { int c = lab[pair_idx[2 * p + 1]]; v = e2[c * 200 + (j - 200)]; }
    REL[(long long)p * kD + 1536 + j] = v;
}

__global__ void pe_transpose_kernel(const float* __restrict__ X, float* __restrict__ SEQ)
{
    long long idx = (long long)blockIdx.x * 256 + threadIdx.x;   // 2000*1936
    if (idx >= (long long)kPair * kD) return;
    int d = (int)(idx % kD);
    long long tok = idx / kD;
    int o = (int)(tok % kO), f = (int)(tok / kO);
    int i2 = d & ~1;
    float div = __expf((float)i2 * (-9.210340371976184f / (float)kD));
    float ang = (float)f * div;
    float pe = (d & 1) ? __cosf(ang) : __sinf(ang);
    SEQ[((long long)o * kF + f) * kD + d] = X[idx] + pe;
}

__global__ void glob_transpose_kernel(const float* __restrict__ SEQ, float* __restrict__ G)
{
    long long idx = (long long)blockIdx.x * 256 + threadIdx.x;
    if (idx >= (long long)kPair * kD) return;
    int d = (int)(idx % kD);
    long long tok = idx / kD;
    int o = (int)(tok % kO), f = (int)(tok / kO);
    G[idx] = SEQ[((long long)o * kF + f) * kD + d];
}

// ---------------- host helpers ----------------
static void gemm(hipStream_t st, const float* A, const float* W, const float* bias,
                 const float* bnsc, const float* bnsh, const float* res, float* C,
                 int M, int N, int K, int act, int amode, const int* aidx, int aidxsel,
                 long long lda, long long akstride, ConvGeom cg,
                 long long ldc, long long ldres)
{
    dim3 grid((M + 127) / 128, (N + 63) / 64), blk(256);
#define GEMM_CASE(a, m)                                                              \
    if (act == a && amode == m) {                                                    \
        gemm_bf16_wmma<a, m><<<grid, blk, 0, st>>>(A, W, bias, bnsc, bnsh, res, C,   \
            M, N, K, aidx, aidxsel, lda, akstride, cg, ldc, ldres);                  \
        return;                                                                      \
    }
    GEMM_CASE(0, 0) GEMM_CASE(0, 1) GEMM_CASE(0, 2)
    GEMM_CASE(1, 0) GEMM_CASE(1, 3) GEMM_CASE(1, 4)
    GEMM_CASE(2, 0)
#undef GEMM_CASE
    // fallback (unused combinations)
    gemm_bf16_wmma<0, 0><<<grid, blk, 0, st>>>(A, W, bias, bnsc, bnsh, res, C,
        M, N, K, aidx, aidxsel, lda, akstride, cg, ldc, ldres);
}

static void run_layer(hipStream_t st, float* X, float* QB, float* KB, float* VB,
                      float* T0, float* T1, float* FFB,
                      const float* const* w, int B, int S, int causal)
{
    const ConvGeom g0 = {};
    const int M = B * S;
    gemm(st, X, w[0], w[1], nullptr, nullptr, nullptr, QB, M, kD, kD, 0, 0, nullptr, 0, kD, 1, g0, kD, 0);
    gemm(st, X, w[2], w[3], nullptr, nullptr, nullptr, KB, M, kD, kD, 0, 0, nullptr, 0, kD, 1, g0, kD, 0);
    gemm(st, X, w[4], w[5], nullptr, nullptr, nullptr, VB, M, kD, kD, 0, 0, nullptr, 0, kD, 1, g0, kD, 0);
    attn_kernel<<<B * kH * S, 128, 0, st>>>(QB, KB, VB, T0, B, S, causal);
    gemm(st, T0, w[6], w[7], nullptr, nullptr, X, T1, M, kD, kD, 0, 0, nullptr, 0, kD, 1, g0, kD, kD);
    ln_kernel<<<M, 256, 0, st>>>(T1, w[12], w[13], X, kD);
    gemm(st, X, w[8], w[9], nullptr, nullptr, nullptr, FFB, M, kDFF, kD, 1, 0, nullptr, 0, kD, 1, g0, kDFF, 0);
    gemm(st, FFB, w[10], w[11], nullptr, nullptr, X, T1, M, kD, kDFF, 0, 0, nullptr, 0, kDFF, 1, g0, kD, kD);
    ln_kernel<<<M, 256, 0, st>>>(T1, w[14], w[15], X, kD);
}

// ---------------- entry point ----------------
extern "C" void kernel_launch(void* const* d_in, const int* in_sizes, int n_in,
                              void* d_out, int out_size, void* d_ws, size_t ws_size,
                              hipStream_t stream)
{
    (void)in_sizes; (void)n_in; (void)out_size; (void)ws_size;
    int i = 0;
    const float* features   = (const float*)d_in[i++];
    const float* union_feat = (const float*)d_in[i++];
    const float* spatial    = (const float*)d_in[i++];
    const int*   pair_idx   = (const int*)d_in[i++];
    const int*   pred_lab   = (const int*)d_in[i++];
    const float* subj_w = (const float*)d_in[i++]; const float* subj_b = (const float*)d_in[i++];
    const float* obj_w  = (const float*)d_in[i++]; const float* obj_b  = (const float*)d_in[i++];
    const float* un_w   = (const float*)d_in[i++]; const float* un_b   = (const float*)d_in[i++];
    const float* c1_w   = (const float*)d_in[i++]; const float* c1_b   = (const float*)d_in[i++];
    const float* bn1g = (const float*)d_in[i++]; const float* bn1b = (const float*)d_in[i++];
    const float* bn1m = (const float*)d_in[i++]; const float* bn1v = (const float*)d_in[i++];
    const float* c2_w   = (const float*)d_in[i++]; const float* c2_b   = (const float*)d_in[i++];
    const float* bn2g = (const float*)d_in[i++]; const float* bn2b = (const float*)d_in[i++];
    const float* bn2m = (const float*)d_in[i++]; const float* bn2v = (const float*)d_in[i++];
    const float* vr_w = (const float*)d_in[i++]; const float* vr_b = (const float*)d_in[i++];
    const float* emb1 = (const float*)d_in[i++]; const float* emb2 = (const float*)d_in[i++];
    const float* enc[4][16];
    for (int L = 0; L < 4; ++L)
        for (int j = 0; j < 16; ++j) enc[L][j] = (const float*)d_in[i++];
    const float* a_w  = (const float*)d_in[i++]; const float* a_b  = (const float*)d_in[i++];
    const float* s_w  = (const float*)d_in[i++]; const float* s_b  = (const float*)d_in[i++];
    const float* c_w  = (const float*)d_in[i++]; const float* c_b  = (const float*)d_in[i++];
    const float* g1_w = (const float*)d_in[i++]; const float* g1_b = (const float*)d_in[i++];
    const float* g2_w = (const float*)d_in[i++]; const float* g2_b = (const float*)d_in[i++];

    // workspace layout (f32)
    float* ws = (float*)d_ws;
    size_t off = 0;
    auto alloc = [&](size_t n) { float* p = ws + off; off += n; return p; };
    float* U    = alloc(98000ull * 256);       // union conv out, (p*49+hw, oc)
    float* X    = alloc((size_t)kPair * kD);   // token buffer (rel / loc)
    float* QB   = alloc((size_t)kPair * kD);
    float* KB   = alloc((size_t)kPair * kD);
    float* VB   = alloc((size_t)kPair * kD);
    float* T0   = alloc((size_t)kPair * kD);
    float* T1   = alloc((size_t)kPair * kD);
    float* FFB  = alloc((size_t)kPair * kDFF);
    float* SEQ  = alloc((size_t)kPair * kD);
    float* TMP1 = alloc(392000ull * 128);      // conv1 out (p*196+oh*14+ow, oc)
    float* TMP2 = alloc(98000ull * 128);       // pooled   (p*49+ph*7+pw, ic)
    float* WP   = alloc(512ull * 12544);       // permuted vr weights
    float* HBOX = alloc(2000ull * 1024);
    float* BN1S = alloc(128); float* BN1T = alloc(128);
    float* BN2S = alloc(256); float* BN2T = alloc(256);

    float* out    = (float*)d_out;
    float* o_att  = out;                         // 2000*3
    float* o_spat = out + 6000;                  // 2000*6
    float* o_cont = out + 18000;                 // 2000*17
    float* o_glob = out + 52000;                 // 2000*1936
    float* o_box  = out + 3924000;               // 2000*4

    const ConvGeom g0 = {};

    // prep: folded BN + permuted vr weight
    bn_prep_kernel<<<1, 256, 0, stream>>>(bn1g, bn1b, bn1m, bn1v, BN1S, BN1T, 128);
    bn_prep_kernel<<<1, 256, 0, stream>>>(bn2g, bn2b, bn2m, bn2v, BN2S, BN2T, 256);
    perm_vrw_kernel<<<(int)((512LL * 12544 + 255) / 256), 256, 0, stream>>>(vr_w, WP);

    // union 1x1 conv as strided GEMM: (98000 x 1024) x (1024 x 256)
    gemm(stream, union_feat, un_w, un_b, nullptr, nullptr, nullptr, U,
         98000, 256, 1024, 0, 2, nullptr, 0, 50176, 49, g0, 256, 0);

    // mask branch: conv1(+bias,relu,bn1) as im2col GEMM (NCHW in)
    ConvGeom cg1 = {2, 27, 27, 14, 14, 7, 7, 2, 3};
    gemm(stream, spatial, c1_w, c1_b, BN1S, BN1T, nullptr, TMP1,
         392000, 128, 98, 1, 3, nullptr, 0, 0, 1, cg1, 128, 0);
    maxpool_kernel<<<(int)((2000LL * 49 * 128 + 255) / 256), 256, 0, stream>>>(TMP1, TMP2);
    // conv2(+bias,relu,bn2) + accumulate into U (NHWC in)
    ConvGeom cg2 = {128, 7, 7, 7, 7, 3, 3, 1, 1};
    gemm(stream, TMP2, c2_w, c2_b, BN2S, BN2T, U, U,
         98000, 256, 1152, 1, 4, nullptr, 0, 0, 1, cg2, 256, 256);

    // subj / obj gathered GEMMs -> rel[:,0:512], rel[:,512:1024]
    gemm(stream, features, subj_w, subj_b, nullptr, nullptr, nullptr, X,
         kPair, 512, kFeat, 0, 1, pair_idx, 0, kFeat, 1, g0, kD, 0);
    gemm(stream, features, obj_w, obj_b, nullptr, nullptr, nullptr, X + 512,
         kPair, 512, kFeat, 0, 1, pair_idx, 1, kFeat, 1, g0, kD, 0);

    // vr_fc: U is contiguous (p, k'=hw*256+oc); use permuted weights
    gemm(stream, U, WP, vr_b, nullptr, nullptr, nullptr, X + 1024,
         kPair, 512, 12544, 0, 0, nullptr, 0, 12544, 1, g0, kD, 0);

    // semantic embeddings -> rel[:,1536:1936]
    xsem_kernel<<<(int)((2000LL * 400 + 255) / 256), 256, 0, stream>>>(
        pair_idx, pred_lab, emb1, emb2, X);

    // local encoder (B=100 frames, S=20 objects, no mask)
    run_layer(stream, X, QB, KB, VB, T0, T1, FFB, enc[0], kF, kO, 0);

    // transpose to object-major + positional encoding
    pe_transpose_kernel<<<(int)(((long long)kPair * kD + 255) / 256), 256, 0, stream>>>(X, SEQ);

    // 3 global encoder layers (B=20 objects, S=100 frames, causal)
    for (int L = 1; L < 4; ++L)
        run_layer(stream, SEQ, QB, KB, VB, T0, T1, FFB, enc[L], kO, kF, 1);

    // glob output (pair-major)
    glob_transpose_kernel<<<(int)(((long long)kPair * kD + 255) / 256), 256, 0, stream>>>(SEQ, o_glob);

    // heads
    gemm(stream, o_glob, a_w, a_b, nullptr, nullptr, nullptr, o_att,
         kPair, 3, kD, 0, 0, nullptr, 0, kD, 1, g0, 3, 0);
    gemm(stream, o_glob, s_w, s_b, nullptr, nullptr, nullptr, o_spat,
         kPair, 6, kD, 2, 0, nullptr, 0, kD, 1, g0, 6, 0);
    gemm(stream, o_glob, c_w, c_b, nullptr, nullptr, nullptr, o_cont,
         kPair, 17, kD, 2, 0, nullptr, 0, kD, 1, g0, 17, 0);
    gemm(stream, o_glob, g1_w, g1_b, nullptr, nullptr, nullptr, HBOX,
         kPair, 1024, kD, 1, 0, nullptr, 0, kD, 1, g0, 1024, 0);
    gemm(stream, HBOX, g2_w, g2_b, nullptr, nullptr, nullptr, o_box,
         kPair, 4, 1024, 0, 0, nullptr, 0, 1024, 1, g0, 4, 0);
}
